// RGCNN_model_249108103363
// MI455X (gfx1250) — compile-verified, matrix operand order checked by reading, not run on hardware
//
#include <hip/hip_runtime.h>
#include <math.h>

// ---------------------------------------------------------------------------
// RGCNN forward for MI455X (gfx1250, wave32, WMMA).
// GEMM operands stored as f16 in memory (An = 32MB -> L2-resident; halves HBM
// traffic), f32 accumulate via v_wmma_f32_16x16x32_f16. Block tile 128x64,
// 8 waves, 32x32 per wave (4 WMMAs per K-chunk). Double-buffered LDS with
// async global->LDS DMA (ASYNCcnt-pipelined): chunk i+1's copies are issued
// before chunk i's WMMAs, synchronized with s_wait_asynccnt <= pending.
// Workspace requirement: ~440 MB (see layout in kernel_launch).
// ---------------------------------------------------------------------------

typedef __attribute__((ext_vector_type(16))) _Float16 v16h;
typedef __attribute__((ext_vector_type(8)))  _Float16 v8h;
typedef __attribute__((ext_vector_type(4)))  _Float16 v4h;
typedef __attribute__((ext_vector_type(8)))  float    v8f;
typedef __attribute__((ext_vector_type(4)))  float    v4f;
typedef __attribute__((ext_vector_type(4)))  int      v4i;

#define BM  128
#define BN  64
#define LDK 40   // LDS row stride in halfs (80B rows, 16B aligned, bank-spread)

#if __has_builtin(__builtin_amdgcn_global_load_async_to_lds_b128)
#define HAVE_ASYNC_LDS 1
typedef v4i __attribute__((address_space(1))) GV4i;   // global int4
typedef v4i __attribute__((address_space(3))) LV4i;   // LDS int4
__device__ __forceinline__ void copy16_to_lds(_Float16* lds_dst, const _Float16* gsrc) {
    __builtin_amdgcn_global_load_async_to_lds_b128((GV4i*)gsrc, (LV4i*)lds_dst, 0, 0);
}
#if __has_builtin(__builtin_amdgcn_s_wait_asynccnt)
#define WAIT_ASYNCCNT(imm) __builtin_amdgcn_s_wait_asynccnt(imm)
#else
#define WAIT_ASYNCCNT(imm) asm volatile("s_wait_asynccnt %0" :: "i"(imm) : "memory")
#endif
// Wait until at most `n` of this wave's async copies remain outstanding.
__device__ __forceinline__ void async_wait_upto(int n) {
    if (n <= 0)      { WAIT_ASYNCCNT(0); }
    else if (n == 1) { WAIT_ASYNCCNT(1); }
    else if (n == 2) { WAIT_ASYNCCNT(2); }
    else             { WAIT_ASYNCCNT(3); }
}
#else
#define HAVE_ASYNC_LDS 0
__device__ __forceinline__ void copy16_to_lds(_Float16* lds_dst, const _Float16* gsrc) {
    *(v8h*)lds_dst = *(const v8h*)gsrc;
}
__device__ __forceinline__ void async_wait_upto(int) {}
#endif

// Load a 16x32 f16 fragment from an LDS tile stored row-major [rows][LDK].
// ISA 7.12.2: lane l<16 -> row M=l, K=0..7 / 16..23;  lane l>=16 -> K=8..15 / 24..31.
__device__ __forceinline__ v16h load_frag(const _Float16* tile, int row0) {
    const int lane = threadIdx.x & 31;
    const int r  = row0 + (lane & 15);
    const int kh = (lane >> 4) << 3;           // 0 or 8
    union { v16h v; struct { v8h lo; v8h hi; } p; } f;
    f.p.lo = *(const v8h*)(tile + r * LDK + kh);
    f.p.hi = *(const v8h*)(tile + r * LDK + kh + 16);
    return f.v;
}

// C[b] = alpha * op(A[b]) @ op(B[b]) + beta * Y[b] + bias ; optional f16 mirror.
//   transA=0: A row-major [M x K];  transA=1: A row-major [K x M]
//   transB=0: B row-major [K x Nn]; transB=1: B row-major [Nn x K]
__global__ __launch_bounds__(256) void wmma_gemm_kernel(
    const _Float16* __restrict__ A, const _Float16* __restrict__ Bm,
    float* __restrict__ C, _Float16* __restrict__ C16,
    int M, int K, int Nn, long sA, long sB, long sC,
    int transA, int transB, float alpha,
    const float* __restrict__ Y, long sY, float beta,
    const float* __restrict__ bias)
{
    __shared__ __align__(16) _Float16 la[2][BM * LDK];
    __shared__ __align__(16) _Float16 lb[2][BN * LDK];

    const int bz = blockIdx.z;
    A += (long)bz * sA; Bm += (long)bz * sB; C += (long)bz * sC;
    if (C16) C16 += (long)bz * sC;
    if (Y)   Y   += (long)bz * sY;

    const int m0 = blockIdx.x * BM, n0 = blockIdx.y * BN;
    const int tid = threadIdx.x, lane = tid & 31, wid = tid >> 5;
    const int wm = (wid & 3) << 5;   // wave rows: 0,32,64,96
    const int wn = (wid >> 2) << 5;  // wave cols: 0,32

    // Fast (unguarded, 16B-vector) staging for interior tiles; the contiguous
    // dimension must be a multiple of 8 halfs for alignment.
    const bool fastA = (m0 + BM <= M) && (((transA ? M : K) & 7) == 0);
    const bool fastB = (n0 + BN <= Nn) && (((transB ? K : Nn) & 7) == 0);
    const int kfull = K & ~31;
    const int nfullc = kfull >> 5;
    const int nch = nfullc + ((kfull < K) ? 1 : 0);

    v8f a00 = {0.f,0.f,0.f,0.f,0.f,0.f,0.f,0.f};
    v8f a01 = a00, a10 = a00, a11 = a00;

    // Stage chunk c into LDS buffer bi. Returns #async copies issued/thread.
    auto stage_chunk = [&](int c, int bi) -> int {
        const int kc = c << 5;
        const int klim = (c < nfullc) ? 32 : (K - kfull);
        _Float16* ta = la[bi];
        _Float16* tb = lb[bi];
        int pend = 0;
        // ---- A tile: ta[r][c] = opA(m0+r, kc+c), zero-padded ----
        if (fastA && klim == 32) {
            if (!transA) {                              // rows contiguous: async DMA
#pragma unroll
                for (int j = 0; j < 2; ++j) {           // 128x32 halfs, 2x16B/thread
                    int slot = j * 256 + tid;
                    int r = slot >> 2, cc = (slot & 3) << 3;
                    copy16_to_lds(ta + r * LDK + cc, A + (long)(m0 + r) * K + kc + cc);
                }
                pend += 2;
            } else {                                    // contiguous along M: transpose in regs
#pragma unroll
                for (int j = 0; j < 2; ++j) {
                    int slot = j * 256 + tid;
                    int k = slot >> 4, m = (slot & 15) << 3;
                    v8h u = *(const v8h*)(A + (long)(kc + k) * M + m0 + m);
#pragma unroll
                    for (int i = 0; i < 8; ++i) ta[(m + i) * LDK + k] = u[i];
                }
            }
        } else {
#pragma unroll
            for (int j = 0; j < 16; ++j) {
                int e = j * 256 + tid;
                int r = e >> 5, cc = e & 31;
                int gm = m0 + r, gk = kc + cc;
                _Float16 v = (_Float16)0.f;
                if (gm < M && cc < klim)
                    v = transA ? A[(long)gk * M + gm] : A[(long)gm * K + gk];
                ta[r * LDK + cc] = v;
            }
        }
        // ---- B tile: tb[n][k] = opB(kc+k, n0+n), zero-padded ----
        if (fastB && klim == 32) {
            if (transB) {                               // rows contiguous: async DMA
                int r = tid >> 2, cc = (tid & 3) << 3;  // 64x32 halfs, 16B/thread
                copy16_to_lds(tb + r * LDK + cc, Bm + (long)(n0 + r) * K + kc + cc);
                pend += 1;
            } else {                                    // contiguous along Nn
                int k = tid >> 3, n = (tid & 7) << 3;
                v8h u = *(const v8h*)(Bm + (long)(kc + k) * Nn + n0 + n);
#pragma unroll
                for (int i = 0; i < 8; ++i) tb[(n + i) * LDK + k] = u[i];
            }
        } else {
#pragma unroll
            for (int j = 0; j < 8; ++j) {
                int e = j * 256 + tid;
                int r = e >> 5, cc = e & 31;
                int gn = n0 + r, gk = kc + cc;
                _Float16 v = (_Float16)0.f;
                if (gn < Nn && cc < klim)
                    v = transB ? Bm[(long)gn * K + gk] : Bm[(long)gk * Nn + gn];
                tb[r * LDK + cc] = v;
            }
        }
        return pend;
    };

    auto compute_chunk = [&](int bi) {
        const _Float16* ta = la[bi];
        const _Float16* tb = lb[bi];
        v16h af0 = load_frag(ta, wm);
        v16h af1 = load_frag(ta, wm + 16);
        v16h bf0 = load_frag(tb, wn);
        v16h bf1 = load_frag(tb, wn + 16);
        a00 = __builtin_amdgcn_wmma_f32_16x16x32_f16(false, af0, false, bf0,
                                                     (short)0, a00, false, false);
        a01 = __builtin_amdgcn_wmma_f32_16x16x32_f16(false, af0, false, bf1,
                                                     (short)0, a01, false, false);
        a10 = __builtin_amdgcn_wmma_f32_16x16x32_f16(false, af1, false, bf0,
                                                     (short)0, a10, false, false);
        a11 = __builtin_amdgcn_wmma_f32_16x16x32_f16(false, af1, false, bf1,
                                                     (short)0, a11, false, false);
    };

    // Software pipeline: issue chunk i+1's DMA into the other buffer before
    // computing chunk i; ASYNCcnt completes in order, so waiting <= pendNext
    // guarantees chunk i's copies landed without draining chunk i+1's.
    stage_chunk(0, 0);
    int cur = 0;
    for (int i = 0; i < nch; ++i) {
        int pendNext = 0;
        if (i + 1 < nch) pendNext = stage_chunk(i + 1, cur ^ 1);
        async_wait_upto(pendNext);
        __syncthreads();          // chunk i staged & visible to all waves
        compute_chunk(cur);
        __syncthreads();          // all waves done reading buf `cur`
        cur ^= 1;
    }

    // Epilogue. C/D layout: lane -> col lane%16, vgpr r -> row r + 8*(lane/16).
    auto store_tile = [&](v8f acc, int ro, int co) {
        int gn = n0 + wn + co + (lane & 15);
        int gm0 = m0 + wm + ro + ((lane >> 4) << 3);
        if (gn < Nn) {
#pragma unroll
            for (int r = 0; r < 8; ++r) {
                int gm = gm0 + r;
                if (gm < M) {
                    float v = alpha * acc[r];
                    if (Y)    v += beta * Y[(long)gm * Nn + gn];
                    if (bias) v += bias[gn];
                    C[(long)gm * Nn + gn] = v;
                    if (C16) C16[(long)gm * Nn + gn] = (_Float16)v;
                }
            }
        }
    };
    store_tile(a00, 0, 0);  store_tile(a01, 0, 16);
    store_tile(a10, 16, 0); store_tile(a11, 16, 16);
}

// f32 -> f16 conversion (one-shot for inputs/weights)
__global__ void cvt_f16_kernel(const float* __restrict__ src,
                               _Float16* __restrict__ dst, long n) {
    for (long i = blockIdx.x * (long)blockDim.x + threadIdx.x; i < n;
         i += (long)gridDim.x * blockDim.x)
        dst[i] = (_Float16)src[i];
}

// sq[i] = sum_c f[i][c]^2   (rows = B*N)
__global__ void rowsq_kernel(const float* __restrict__ f, float* __restrict__ sq,
                             int rows, int C) {
    int i = blockIdx.x * blockDim.x + threadIdx.x;
    if (i < rows) {
        const float* p = f + (long)i * C;
        float s = 0.f;
        for (int c = 0; c < C; ++c) s += p[c] * p[c];
        sq[i] = s;
    }
}

// In-place: G -> A = exp(-(sq_i - 2G + sq_j)), zero diag; dinv = rsqrt(rowsum).
__global__ __launch_bounds__(256) void adj_kernel(float* __restrict__ G,
                                                  const float* __restrict__ sq,
                                                  float* __restrict__ dinv) {
    const int i = blockIdx.x, b = blockIdx.y;
    float* row = G + ((long)b * 1024 + i) * 1024;
    const float* sqb = sq + b * 1024;
    const float si = sqb[i];
    const int j = threadIdx.x << 2;
    v4f g = *(const v4f*)(row + j);
    v4f s = *(const v4f*)(sqb + j);
    v4f a;
    float sum = 0.f;
#pragma unroll
    for (int t = 0; t < 4; ++t) {
        float d = si - 2.f * g[t] + s[t];
        float av = __expf(-d);
        if (j + t == i) av = 0.f;
        a[t] = av;
        sum += av;
    }
    *(v4f*)(row + j) = a;
    __shared__ float red[256];
    red[threadIdx.x] = sum; __syncthreads();
    for (int t = 128; t > 0; t >>= 1) {
        if (threadIdx.x < t) red[threadIdx.x] += red[threadIdx.x + t];
        __syncthreads();
    }
    if (threadIdx.x == 0) dinv[b * 1024 + i] = rsqrtf(red[0]);
}

// A16[b,i,j] = (f16)(dinv[b,i] * A[b,i,j] * dinv[b,j])
__global__ void normalize_kernel(const float* __restrict__ A,
                                 const float* __restrict__ dinv,
                                 _Float16* __restrict__ A16) {
    const long total4 = (long)16 * 1024 * 256;
    for (long idx = blockIdx.x * (long)blockDim.x + threadIdx.x; idx < total4;
         idx += (long)gridDim.x * blockDim.x) {
        long b = idx >> 18;
        int r = (int)((idx >> 8) & 1023);
        int c = (int)((idx & 255) << 2);
        v4f v = ((const v4f*)A)[idx];
        const float di = dinv[b * 1024 + r];
        const float* dj = dinv + b * 1024 + c;
        v4h h;
#pragma unroll
        for (int t = 0; t < 4; ++t) h[t] = (_Float16)(v[t] * di * dj[t]);
        ((v4h*)A16)[idx] = h;
    }
}

// relu in place on f32 + f16 mirror
__global__ void relu_cvt_kernel(float* __restrict__ x, _Float16* __restrict__ x16,
                                long n) {
    for (long i = blockIdx.x * (long)blockDim.x + threadIdx.x; i < n;
         i += (long)gridDim.x * blockDim.x) {
        float v = fmaxf(x[i], 0.f);
        x[i] = v;
        x16[i] = (_Float16)v;
    }
}

__global__ __launch_bounds__(256) void sumsq_kernel(const float* __restrict__ x,
                                                    long n, float* __restrict__ slot) {
    float s = 0.f;
    for (long i = blockIdx.x * (long)blockDim.x + threadIdx.x; i < n;
         i += (long)gridDim.x * blockDim.x) { float v = x[i]; s += v * v; }
    __shared__ float red[256];
    red[threadIdx.x] = s; __syncthreads();
    for (int t = 128; t > 0; t >>= 1) {
        if (threadIdx.x < t) red[threadIdx.x] += red[threadIdx.x + t];
        __syncthreads();
    }
    if (threadIdx.x == 0) atomicAdd(slot, red[0]);
}

__global__ void zero_kernel(float* __restrict__ p, int n) {
    int i = blockIdx.x * blockDim.x + threadIdx.x;
    if (i < n) p[i] = 0.f;
}

// pooled[b,c] = max_n out3[b,n,c]   (out3: [16,1024,1024]; coalesced across c)
__global__ void pool_max_kernel(const float* __restrict__ o3, float* __restrict__ pooled) {
    int i = blockIdx.x * blockDim.x + threadIdx.x;
    if (i >= 16 * 1024) return;
    int b = i >> 10, c = i & 1023;
    const float* base = o3 + ((long)b * 1024) * 1024 + c;
    float m = -3.402823466e38f;
    for (int n = 0; n < 1024; ++n) m = fmaxf(m, base[(long)n * 1024]);
    pooled[i] = m;
}

__global__ void fc_kernel(const float* __restrict__ in, const float* __restrict__ W,
                          const float* __restrict__ bias, float* __restrict__ outp,
                          int Bn, int Kin, int Kout, int do_relu) {
    int i = blockIdx.x * blockDim.x + threadIdx.x;
    if (i >= Bn * Kout) return;
    int b = i / Kout, o = i % Kout;
    const float* row = in + (long)b * Kin;
    float s = bias[o];
    for (int k = 0; k < Kin; ++k) s += row[k] * W[(long)k * Kout + o];
    if (do_relu) s = fmaxf(s, 0.f);
    outp[i] = s;
}

// regs order: r1, r2, r3, nw, nb, nw, nb, nw, nb
__global__ void finalize_kernel(const float* __restrict__ slots, float* __restrict__ r) {
    if (threadIdx.x == 0 && blockIdx.x == 0) {
        float r1 = sqrtf(slots[0]), r2 = sqrtf(slots[1]), r3 = sqrtf(slots[2]);
        float nw = sqrtf(slots[3]), nb = sqrtf(slots[4]);
        r[0] = r1; r[1] = r2; r[2] = r3;
        r[3] = nw; r[4] = nb; r[5] = nw; r[6] = nb; r[7] = nw; r[8] = nb;
    }
}

extern "C" void kernel_launch(void* const* d_in, const int* in_sizes, int n_in,
                              void* d_out, int out_size, void* d_ws, size_t ws_size,
                              hipStream_t stream) {
    const float* x    = (const float*)d_in[0];   // [16,1024,6]
    const float* w1   = (const float*)d_in[1];   // [6,6,128]
    const float* b1   = (const float*)d_in[2];
    const float* w2   = (const float*)d_in[3];   // [5,128,512]
    const float* b2   = (const float*)d_in[4];
    const float* w3   = (const float*)d_in[5];   // [3,512,1024]
    const float* b3   = (const float*)d_in[6];
    const float* fc1w = (const float*)d_in[7];   // [1024,512]
    const float* fc1b = (const float*)d_in[8];
    const float* fc2w = (const float*)d_in[9];   // [512,128]
    const float* fc2b = (const float*)d_in[10];
    const float* fc3w = (const float*)d_in[11];  // [128,40]
    const float* fc3b = (const float*)d_in[12];
    float* out = (float*)d_out;                  // 640 logits + 9 regs
    float* ws  = (float*)d_ws;

    const int B = 16, N = 1024;
    const long M1 = 1 << 20;
    const long NN = (long)N * N;

    // Workspace layout (float slots; f16 buffers cast). ~440 MB total.
    float*    ANf  = ws;                          // 16M f32 adjacency scratch
    float*    Pf   = ws;                          //   alias (ANf dead after normalize)
    float*    MMf  = ws;                          //   alias (Pf dead once P16 written)
    _Float16* AN16 = (_Float16*)(ws + 16 * M1);   // 16M halfs
    float*    T0f  = ws + 24 * M1;                // 8M each
    float*    T1f  = ws + 32 * M1;
    float*    T2f  = ws + 40 * M1;
    _Float16* T0h  = (_Float16*)(ws + 48 * M1);   // 8M halfs each
    _Float16* T1h  = (_Float16*)(ws + 52 * M1);
    _Float16* T2h  = (_Float16*)(ws + 56 * M1);
    _Float16* P16  = (_Float16*)(ws + 48 * M1);   // alias T0h/T1h (dead post-cheb)
    float*    OUTAf = ws + 60 * M1;               // 16M each
    float*    OUTBf = ws + 76 * M1;
    _Float16* O16A = (_Float16*)(ws + 92 * M1);   // 16M halfs each
    _Float16* O16B = (_Float16*)(ws + 100 * M1);
    long off = 108 * M1;
    _Float16* X16 = (_Float16*)(ws + off); off += 65536;
    _Float16* W1h = (_Float16*)(ws + off); off += 4096;
    _Float16* W2h = (_Float16*)(ws + off); off += 163840;
    _Float16* W3h = (_Float16*)(ws + off); off += 786432;
    float* SQ   = ws + off; off += 16384;
    float* DINV = ws + off; off += 16384;
    float* POOL = ws + off; off += 16384;
    float* H1   = ws + off; off += 8192;
    float* H2   = ws + off; off += 2048;
    float* SLOT = ws + off;

    zero_kernel<<<1, 8, 0, stream>>>(SLOT, 8);

    // one-shot f16 conversions of raw inputs / weights
    cvt_f16_kernel<<<96, 256, 0, stream>>>(x,  X16, (long)B * N * 6);
    cvt_f16_kernel<<<18, 256, 0, stream>>>(w1, W1h, 6L * 6 * 128);
    cvt_f16_kernel<<<640, 256, 0, stream>>>(w2, W2h, 5L * 128 * 512);
    cvt_f16_kernel<<<2048, 256, 0, stream>>>(w3, W3h, 3L * 512 * 1024);

    auto gemm = [&](const _Float16* A, const _Float16* Bm, float* C, _Float16* C16,
                    int M, int K, int Nn, long sA, long sB, long sC, int tA, int tB,
                    float alpha, const float* Y, long sY, float beta,
                    const float* bias, int batch) {
        dim3 g((M + BM - 1) / BM, (Nn + BN - 1) / BN, batch);
        wmma_gemm_kernel<<<g, 256, 0, stream>>>(A, Bm, C, C16, M, K, Nn, sA, sB, sC,
                                                tA, tB, alpha, Y, sY, beta, bias);
    };
    auto sumsq = [&](const float* p, long n, float* slot) {
        int blocks = (int)((n + 255) / 256); if (blocks > 2048) blocks = 2048;
        sumsq_kernel<<<blocks, 256, 0, stream>>>(p, n, slot);
    };

    // One RGCNN block: graph from feat, ChebConv(Kc terms), ReLU, regularizer.
    auto cheb_block = [&](const _Float16* feat16, const float* featF, int Fin,
                          const _Float16* Wk16, const float* bias, int Kc, int O,
                          float* OUTf, _Float16* OUT16, int slot) {
        const long sF = (long)N * Fin, sO = (long)N * O;
        // -- graph build: Gram (WMMA NT, f16) -> exp/deg -> sym-normalize to f16 --
        rowsq_kernel<<<(B * N + 255) / 256, 256, 0, stream>>>(featF, SQ, B * N, Fin);
        gemm(feat16, feat16, ANf, nullptr, N, Fin, N, sF, sF, NN, 0, 1, 1.f,
             nullptr, 0, 0.f, nullptr, B);
        adj_kernel<<<dim3(N, B), 256, 0, stream>>>(ANf, SQ, DINV);
        normalize_kernel<<<2048, 256, 0, stream>>>(ANf, DINV, AN16);
        // -- ChebConv: OUT = sum_k T_k @ W_k + bias ;  L_hat = -An --
        gemm(feat16, Wk16, OUTf, nullptr, B * N, Fin, O, 0, 0, 0, 0, 0, 1.f,
             nullptr, 0, 0.f, bias, 1);                       // T0 = feat
        gemm(AN16, feat16, T1f, T1h, N, N, Fin, NN, sF, sF, 0, 0, -1.f,
             nullptr, 0, 0.f, nullptr, B);                    // T1 = -An @ feat
        gemm(T1h, Wk16 + (long)Fin * O, OUTf, nullptr, B * N, Fin, O, 0, 0, 0,
             0, 0, 1.f, OUTf, 0, 1.f, nullptr, 1);
        // Chebyshev recurrence: T_k = -2*An@T_{k-1} - T_{k-2}
        // f32 chain (t0F/t1F) feeds the Y-add; f16 chain (t1h) feeds WMMA.
        const float* t0F = featF;
        const float* t1F = T1f;
        const _Float16* t1h = T1h;
        float*    bufF[3] = {T2f, T0f, T1f};
        _Float16* bufH[3] = {T2h, T0h, T1h};
        int sel = 0;
        for (int k = 2; k < Kc; ++k) {
            float* dF = bufF[sel]; _Float16* dH = bufH[sel]; sel = (sel + 1) % 3;
            gemm(AN16, t1h, dF, dH, N, N, Fin, NN, sF, sF, 0, 0, -2.f,
                 t0F, sF, -1.f, nullptr, B);
            gemm(dH, Wk16 + (long)k * Fin * O, OUTf, nullptr, B * N, Fin, O,
                 0, 0, 0, 0, 0, 1.f, OUTf, 0, 1.f, nullptr, 1);
            t0F = t1F; t1F = dF; t1h = dH;
        }
        relu_cvt_kernel<<<2048, 256, 0, stream>>>(OUTf, OUT16, (long)B * N * O);
        // -- regularizer: ||out^T (I - An) out||_F --
        gemm(AN16, OUT16, Pf, P16, N, N, O, NN, sO, sO, 0, 0, -1.f,
             OUTf, sO, 1.f, nullptr, B);                      // P = out - An@out
        gemm(OUT16, P16, MMf, nullptr, O, N, O, sO, sO, (long)O * O, 1, 0, 1.f,
             nullptr, 0, 0.f, nullptr, B);                    // M = out^T @ P
        sumsq(MMf, (long)B * O * O, SLOT + slot);
    };

    cheb_block(X16,  x,     6,   W1h, b1, 6, 128,  OUTAf, O16A, 0);
    cheb_block(O16A, OUTAf, 128, W2h, b2, 5, 512,  OUTBf, O16B, 1);
    cheb_block(O16B, OUTBf, 512, W3h, b3, 3, 1024, OUTAf, O16A, 2);

    // -- head: max pool + FC stack (tiny; plain kernels) --
    pool_max_kernel<<<(16 * 1024 + 255) / 256, 256, 0, stream>>>(OUTAf, POOL);
    fc_kernel<<<(16 * 512 + 255) / 256, 256, 0, stream>>>(POOL, fc1w, fc1b, H1,
                                                          16, 1024, 512, 1);
    fc_kernel<<<(16 * 128 + 255) / 256, 256, 0, stream>>>(H1, fc2w, fc2b, H2,
                                                          16, 512, 128, 1);
    fc_kernel<<<(16 * 40 + 255) / 256, 256, 0, stream>>>(H2, fc3w, fc3b, out,
                                                         16, 128, 40, 0);
    sumsq(fc1w, (long)1024 * 512, SLOT + 3);
    sumsq(fc1b, 512, SLOT + 4);
    finalize_kernel<<<1, 32, 0, stream>>>(SLOT, out + 640);
}
